// LieGroupRotationPDC_58952721105019
// MI455X (gfx1250) — matrix operand order repaired and checked by reading, not compiled, not run on hardware
//
#include <hip/hip_runtime.h>
#include <cmath>

// ---------------------------------------------------------------------------
// Fused "LieGroupRotationPDC" = single 5x5 conv with effective kernel:
//   K_eff[o,ci,h,w] = mean_over_8_rotations( weight*alpha*PHT ) - 0.7*kd[o,ci]*[h==2&&w==2]
// Implicit GEMM with v_wmma_f32_16x16x32_f16 (wave32, gfx1250).
//   M = 64 (Co), N = 32*128*128 (b,y,x), K = 1600 (kh*320 + kw*64 + ci)
// ---------------------------------------------------------------------------

typedef __attribute__((ext_vector_type(16))) _Float16 v16h;
typedef __attribute__((ext_vector_type(8)))  _Float16 v8h;
typedef __attribute__((ext_vector_type(2)))  _Float16 h2;
typedef __attribute__((ext_vector_type(8)))  float    v8f;

#define CI        64
#define CO        64
#define HW        128
#define NB        32
#define KBLKS     50            // 1600 / 32
#define SM_STRIDE 72            // halves per (r,c) cell: 64 data + 8 pad (16B align)
#define TILE_C    132           // 128 + 4 halo columns
#define ROW_HALVES (TILE_C * SM_STRIDE)

// ---------------------------------------------------------------------------
// Prep: build K_eff in f16, scattered directly into WMMA A-fragment order.
//   A-frag (16-bit A 16x32): lane L -> M = L%16, kbase=(L/16)*8,
//   element e<8 -> K=kbase+e ; e>=8 -> K=16+kbase+(e-8).
//   wswz half index = (((kblk*4 + mo)*32 + lane)*16 + e)
// ---------------------------------------------------------------------------
__global__ void pht_prep_kernel(const float* __restrict__ weight,
                                const float* __restrict__ alpha,
                                _Float16* __restrict__ wswz) {
  const double PI_D = 3.14159265358979323846;
  int t = blockIdx.x * blockDim.x + threadIdx.x;
  if (t >= CO * CI) return;
  int o  = t >> 6;
  int ci = t & 63;
  float al = alpha[0];

  float w[25], weff[25], accum[25];
  float kd = 0.f;
  const float* wp = weight + (size_t)(o * CI + ci) * 25;
#pragma unroll
  for (int i = 0; i < 25; ++i) { w[i] = wp[i]; kd += w[i]; }

  // PHT kernel (faithful: H = cos(4*pi*r2), sw = target/H, pht = H*sw + eps)
#pragma unroll
  for (int h = 0; h < 5; ++h) {
#pragma unroll
    for (int c = 0; c < 5; ++c) {
      float gu = (float)(c - 2), gv = (float)(h - 2);
      float r2 = (gu * gu + gv * gv) / (4.0f + 1e-6f);
      float H  = cosf(4.0f * (float)PI_D * r2);
      float tgt = 0.f;
      if (h == 2 && c == 2) tgt = -4.f;
      else if ((h == 1 && c == 2) || (h == 3 && c == 2) ||
               (h == 2 && c == 1) || (h == 2 && c == 3)) tgt = 1.f;
      float sw  = (tgt != 0.f) ? tgt / H : 0.f;
      float pht = H * sw + 1e-6f;
      weff[h * 5 + c] = w[h * 5 + c] * al * pht;
      accum[h * 5 + c] = 0.f;
    }
  }

  // 8 SO(2) rotations: float32 trig, double products, int() truncation,
  // last-(h,w)-writer-wins (matches reference map construction exactly).
  for (int ai = 0; ai < 8; ++ai) {
    float af = (float)((double)ai * (PI_D / 4.0));
    double cc = (double)cosf(af);
    double ss = (double)sinf(af);
    float rot[25];
#pragma unroll
    for (int i = 0; i < 25; ++i) rot[i] = 0.f;
    for (int h = 0; h < 5; ++h) {
      for (int c = 0; c < 5; ++c) {
        int rh = (int)((double)h * cc + (double)c * (-ss));
        int rw = (int)((double)h * ss + (double)c * cc);
        if (rh >= 0 && rh < 5 && rw >= 0 && rw < 5)
          rot[rh * 5 + rw] = weff[h * 5 + c];
      }
    }
#pragma unroll
    for (int i = 0; i < 25; ++i) accum[i] += rot[i];
  }

  // Scatter into swizzled A-fragment storage.
#pragma unroll
  for (int kh = 0; kh < 5; ++kh) {
#pragma unroll
    for (int kw = 0; kw < 5; ++kw) {
      float v = accum[kh * 5 + kw] * 0.125f;
      if (kh == 2 && kw == 2) v -= 0.7f * kd;          // fold -theta * kd branch
      int k    = kh * 320 + kw * 64 + ci;
      int kblk = k >> 5;
      int kk   = k & 31;
      int lhi  = (kk < 16) ? (kk >> 3) : ((kk - 16) >> 3);
      int e    = (kk < 16) ? (kk & 7) : (8 + (kk & 7));
      int lane = lhi * 16 + (o & 15);
      int mo   = o >> 4;
      size_t idx = ((size_t)((kblk * 4 + mo) * 32 + lane)) * 16 + e;
      wswz[idx] = (_Float16)v;
    }
  }
}

// ---------------------------------------------------------------------------
// Conv: one block per (batch, output row). 8 waves; wave = (mo, ngrp):
//   mo in 0..3 -> output channels mo*16..mo*16+15 ; (ngrp*4+j) -> x tile.
// ---------------------------------------------------------------------------
__global__ __launch_bounds__(256) void pdc_conv_kernel(
    const float* __restrict__ x,
    const _Float16* __restrict__ wswz,
    float* __restrict__ out) {
  __shared__ __align__(16) _Float16 smem[5 * ROW_HALVES];  // 95,040 B

  int tid  = threadIdx.x;
  int lane = tid & 31;
  int wv   = tid >> 5;
  int blk  = blockIdx.x;             // 0 .. 32*128-1
  int b = blk >> 7;
  int y = blk & 127;
  const float* xb = x + (size_t)b * CI * (HW * HW);

  // --- Stage input halo tile: fp32 -> f16, layout [r][c][ci] (ci contiguous).
  //     lane -> column (coalesced 128B global reads), wave -> ci-pair slice.
  for (int r = 0; r < 5; ++r) {
    int yy = y + r - 2;
    bool yok = (unsigned)yy < 128u;
    for (int cp = wv; cp < 32; cp += 8) {
      const float* p0 = xb + (size_t)(cp * 2) * (HW * HW) + yy * HW;
#pragma unroll
      for (int c = lane; c < TILE_C; c += 32) {
        int xx = c - 2;
        float f0 = 0.f, f1 = 0.f;
        if (yok && (unsigned)xx < 128u) {
          f0 = p0[xx];
          f1 = p0[xx + HW * HW];
        }
        h2 pk = {(_Float16)f0, (_Float16)f1};
        *(h2*)(&smem[(r * TILE_C + c) * SM_STRIDE + cp * 2]) = pk;
      }
    }
  }
  __syncthreads();

  // --- Implicit GEMM with WMMA ---
  int mo   = wv >> 1;                // 0..3  (M tile)
  int ngrp = wv & 1;                 // 0..1  (N tile group of 4)
  int lo16 = lane & 15;
  int lhi  = lane >> 4;

  v8f zero = {0.f, 0.f, 0.f, 0.f, 0.f, 0.f, 0.f, 0.f};
  v8f acc0 = zero, acc1 = zero, acc2 = zero, acc3 = zero;

  const v16h* ap = (const v16h*)wswz;   // one v16h = one lane's A-frag slice
  int aidx = mo * 32 + lane;            // kblk 0
  v16h afrag = ap[aidx];

  // Lane-constant pieces of the B address:
  //   off = kh*ROW_HALVES + ((ngrp*64 + lo16 + kw) + j*16)*SM_STRIDE + half*32 + lhi*8
  int colbase = (ngrp * 64 + lo16) * SM_STRIDE + lhi * 8;
  int rowbase = 0;                      // kh * ROW_HALVES, carried
  int kw = 0;                           // carried, no division anywhere

  for (int t25 = 0; t25 < 25; ++t25) {
    int cbase = rowbase + colbase + kw * SM_STRIDE;
#pragma unroll
    for (int half = 0; half < 2; ++half) {
      // Software-pipelined A fragment (clamped index: uniform, branch-free).
      int anext_idx = aidx + 4 * 32;
      if (t25 == 24 && half == 1) anext_idx = aidx;
      v16h anext = ap[anext_idx];
      __builtin_prefetch(&ap[(anext_idx + 4 * 32 <= (KBLKS - 1) * 4 * 32 + 127)
                                 ? anext_idx + 4 * 32 : anext_idx], 0, 3);

      int cb = cbase + (half << 5);     // +32 halves for second ci half
      // Issue all 8 ds_load_b128 before any WMMA.
      v16h bf0, bf1, bf2, bf3;
      {
        int o0 = cb;
        int o1 = cb + 16 * SM_STRIDE;
        int o2 = cb + 32 * SM_STRIDE;
        int o3 = cb + 48 * SM_STRIDE;
        v8h l0 = *(const v8h*)(&smem[o0]), h0 = *(const v8h*)(&smem[o0 + 16]);
        v8h l1 = *(const v8h*)(&smem[o1]), h1 = *(const v8h*)(&smem[o1 + 16]);
        v8h l2 = *(const v8h*)(&smem[o2]), h2v = *(const v8h*)(&smem[o2 + 16]);
        v8h l3 = *(const v8h*)(&smem[o3]), h3 = *(const v8h*)(&smem[o3 + 16]);
        bf0 = __builtin_shufflevector(l0, h0, 0,1,2,3,4,5,6,7,8,9,10,11,12,13,14,15);
        bf1 = __builtin_shufflevector(l1, h1, 0,1,2,3,4,5,6,7,8,9,10,11,12,13,14,15);
        bf2 = __builtin_shufflevector(l2, h2v, 0,1,2,3,4,5,6,7,8,9,10,11,12,13,14,15);
        bf3 = __builtin_shufflevector(l3, h3, 0,1,2,3,4,5,6,7,8,9,10,11,12,13,14,15);
      }
      acc0 = __builtin_amdgcn_wmma_f32_16x16x32_f16(false, afrag, false, bf0,
                                                    (short)0, acc0, false, false);
      acc1 = __builtin_amdgcn_wmma_f32_16x16x32_f16(false, afrag, false, bf1,
                                                    (short)0, acc1, false, false);
      acc2 = __builtin_amdgcn_wmma_f32_16x16x32_f16(false, afrag, false, bf2,
                                                    (short)0, acc2, false, false);
      acc3 = __builtin_amdgcn_wmma_f32_16x16x32_f16(false, afrag, false, bf3,
                                                    (short)0, acc3, false, false);
      afrag = anext;
      aidx  = anext_idx;
    }
    // Carried (kh, kw) counters: no division in the hot loop.
    ++kw;
    if (kw == 5) { kw = 0; rowbase += ROW_HALVES; }
  }

  // --- Epilogue: D layout VGPR r -> M = r + 8*(lane/16), N = lane%16 ---
  float* ob = out + (size_t)b * CO * (HW * HW) + (size_t)y * HW;
  v8f accs[4] = {acc0, acc1, acc2, acc3};
#pragma unroll
  for (int j = 0; j < 4; ++j) {
    int xcol = (ngrp * 4 + j) * 16 + lo16;
#pragma unroll
    for (int r = 0; r < 8; ++r) {
      int o = mo * 16 + 8 * lhi + r;
      ob[(size_t)o * (HW * HW) + xcol] = accs[j][r];
    }
  }
}

// ---------------------------------------------------------------------------
extern "C" void kernel_launch(void* const* d_in, const int* in_sizes, int n_in,
                              void* d_out, int out_size, void* d_ws, size_t ws_size,
                              hipStream_t stream) {
  (void)in_sizes; (void)n_in; (void)out_size; (void)ws_size;
  const float* x      = (const float*)d_in[0];
  const float* weight = (const float*)d_in[1];
  const float* alpha  = (const float*)d_in[2];
  float* out          = (float*)d_out;
  _Float16* wswz      = (_Float16*)d_ws;     // 204,800 B of scratch

  pht_prep_kernel<<<(CO * CI + 255) / 256, 256, 0, stream>>>(weight, alpha, wswz);
  pdc_conv_kernel<<<NB * HW, 256, 0, stream>>>(x, wswz, out);
}
